// AdaptiveResidualFeatureRefinement_1039382086379
// MI455X (gfx1250) — matrix operand order repaired and checked by reading, not compiled, not run on hardware
//
#include <hip/hip_runtime.h>
#include <hip/hip_bf16.h>

typedef __bf16 bf16;
typedef bf16  v16bf __attribute__((ext_vector_type(16)));
typedef bf16  v8bf  __attribute__((ext_vector_type(8)));
typedef float v8f   __attribute__((ext_vector_type(8)));

#define C_CH 256
#define Hdim 128
#define Wdim 128
#define HWp  (Hdim * Wdim)      // 16384 = 2^14
#define Bn   8
#define NPIX (Bn * HWp)         // 131072 pixels

union AFrag { v16bf v; v8bf h[2]; };

// ---------------------------------------------------------------------------
// Layout converters / weight packers
// ---------------------------------------------------------------------------

// x: f32 NCHW -> bf16 NHWC
__global__ __launch_bounds__(256)
void k_nchw_to_nhwc(const float* __restrict__ x, bf16* __restrict__ out) {
    unsigned i = blockIdx.x * 256u + threadIdx.x;          // over B*C*H*W
    unsigned w = i & (Wdim - 1);
    unsigned h = (i >> 7) & (Hdim - 1);
    unsigned c = (i >> 14) & (C_CH - 1);
    unsigned b = i >> 22;
    size_t o = ((size_t)(b * HWp + h * Wdim + w) << 8) + c;
    out[o] = (bf16)x[i];
}

// w: f32 OIHW [256][256][3][3] -> bf16 [tap][cout][cin]
__global__ __launch_bounds__(256)
void k_pack_w3(const float* __restrict__ w, bf16* __restrict__ out) {
    unsigned i = blockIdx.x * 256u + threadIdx.x;          // over 256*256*9
    unsigned tap = i % 9u;
    unsigned ci  = (i / 9u) & (C_CH - 1);
    unsigned co  = i / (9u * C_CH);
    out[(size_t)tap * (C_CH * C_CH) + co * C_CH + ci] = (bf16)w[i];
}

// 1x1 weights: OI layout already matches [cout][cin]; just convert dtype
__global__ __launch_bounds__(256)
void k_pack_w1(const float* __restrict__ w, bf16* __restrict__ out) {
    unsigned i = blockIdx.x * 256u + threadIdx.x;          // over 256*256
    out[i] = (bf16)w[i];
}

// ---------------------------------------------------------------------------
// Implicit-GEMM conv via WMMA bf16 -> f32 with LDS-staged weights.
//   Wave tile : 16 pixels x 256 couts (16 x v8f accumulators)
//   Block     : 8 waves = 128 consecutive pixels (one image row), all couts
//   B staging : LDS tile [256 cout][128 cin] (64 KB), two halves per tap
//   Pipelining: B double-buffered from LDS, A prefetched across K-chunks,
//               WMMA operand-reuse hint (RA) on the shared A fragment.
// out_mode: 0 = bf16 NHWC (outh)
//           1 = f32 NCHW  (outf)
//           2 = f32 NCHW, store only where cmap <= 0.5 (routing fuse)
// ---------------------------------------------------------------------------
__global__ __launch_bounds__(256)
void k_conv_gemm(const bf16* __restrict__ in,
                 const bf16* __restrict__ wpk,
                 const float* __restrict__ bias,
                 const bf16* __restrict__ resid,
                 const float* __restrict__ cmap,
                 float* __restrict__ outf,
                 bf16* __restrict__ outh,
                 int ntaps, int dil, int out_mode) {
    __shared__ bf16 Bs[256 * 128];                 // 64 KB weight tile

    const int tid  = threadIdx.x;
    const int lane = tid & 31;
    const int wid  = tid >> 5;
    const int pixel_base = blockIdx.x * 128 + wid * 16;

    // A lane mapping (16-bit A 16x32): lane&15 = M row, lane>>4 = K half.
    // B lane mapping (16-bit B 32x16): lane&15 = N col, lane>>4 = K half.
    const int lm  = lane & 15;
    const int sel = lane >> 4;
    const int p   = pixel_base + lm;
    const int b   = p >> 14;
    const int rem = p & (HWp - 1);
    const int h   = rem >> 7;
    const int w   = rem & (Wdim - 1);

    // cooperative-fill indices: round r covers couts r*16..r*16+15
    const int f_co = tid >> 4;                     // 0..15 within a round
    const int f_ci = (tid & 15) * 8;               // 0..120 step 8

    v8bf zb;
#pragma unroll
    for (int i = 0; i < 8; ++i) zb[i] = (bf16)0.0f;

    v8f acc[16];
#pragma unroll
    for (int t = 0; t < 16; ++t)
#pragma unroll
        for (int r = 0; r < 8; ++r) acc[t][r] = 0.0f;

    // LDS read base for B fragments of this lane
    const bf16* bs_lane = Bs + lm * 128 + sel * 16;

    for (int tap = 0; tap < ntaps; ++tap) {
        int dy = 0, dx = 0;
        if (ntaps != 1) {
            dy = dil * (tap / 3 - 1);
            dx = dil * (tap % 3 - 1);
        }
        const int sh = h + dy, sw = w + dx;
        const bool valid = (sh >= 0) & (sh < Hdim) & (sw >= 0) & (sw < Wdim);
        const bf16* arow = in + ((size_t)(b * HWp + sh * Wdim + sw)) * C_CH;
        const bf16* wtap = wpk + (size_t)tap * (C_CH * C_CH);

        for (int half = 0; half < 2; ++half) {
            __syncthreads();                       // protect previous readers
            // Fill Bs[co][0..127] = wtap[co][half*128 .. +128): 16 rounds,
            // each round = 256 threads x 16B = 4 KB coalesced.
#pragma unroll
            for (int r = 0; r < 16; ++r) {
                const int co = r * 16 + f_co;
                *(v8bf*)(Bs + co * 128 + f_ci) =
                    *(const v8bf*)(wtap + co * C_CH + half * 128 + f_ci);
            }
            __syncthreads();

            // A prefetch for first K-chunk of this half
            AFrag a_cur;
            {
                const int kc = half * 128;
                if (valid) {
                    a_cur.h[0] = *(const v8bf*)(arow + kc + sel * 16);
                    a_cur.h[1] = *(const v8bf*)(arow + kc + sel * 16 + 8);
                } else {
                    a_cur.h[0] = zb;
                    a_cur.h[1] = zb;
                }
            }

#pragma unroll
            for (int kcl = 0; kcl < 4; ++kcl) {    // 4 x K=32 per half
                // prefetch A for next K-chunk while this chunk computes
                AFrag a_nxt;
                if (kcl < 3) {
                    const int kcn = half * 128 + (kcl + 1) * 32;
                    if (valid) {
                        a_nxt.h[0] = *(const v8bf*)(arow + kcn + sel * 16);
                        a_nxt.h[1] = *(const v8bf*)(arow + kcn + sel * 16 + 8);
                    } else {
                        a_nxt.h[0] = zb;
                        a_nxt.h[1] = zb;
                    }
                }

                // 16 WMMAs sharing one A fragment; B double-buffered from LDS
                const bf16* bk = bs_lane + kcl * 32;
                v16bf bbuf[2];
                bbuf[0] = *(const v16bf*)(bk);                 // t = 0
                bbuf[1] = *(const v16bf*)(bk + 1 * 16 * 128);  // t = 1
                acc[0] = __builtin_amdgcn_wmma_f32_16x16x32_bf16(
                    false, a_cur.v, false, bbuf[0], (short)0, acc[0],
                    false, false);
#pragma unroll
                for (int t = 1; t < 16; ++t) {
                    if (t < 15)
                        bbuf[(t + 1) & 1] =
                            *(const v16bf*)(bk + (t + 1) * 16 * 128);
                    acc[t] = __builtin_amdgcn_wmma_f32_16x16x32_bf16(
                        false, a_cur.v, false, bbuf[t & 1], (short)0, acc[t],
                        /*reuse_a=*/true, /*reuse_b=*/false);
                }

                if (kcl < 3) a_cur = a_nxt;
            }
        }
    }

    // Epilogue: C/D layout — lane&15 = N col; lane>=16 -> rows 8..15
    const int nlane = lane & 15;
    const int mhalf = lane >> 4;
    const bool has_res = (resid != nullptr);
#pragma unroll
    for (int t = 0; t < 16; ++t) {
        const int co = t * 16 + nlane;
        const float bv = bias[co];
#pragma unroll
        for (int r = 0; r < 8; ++r) {
            const int pix = pixel_base + r + mhalf * 8;
            float v = acc[t][r] + bv;
            if (has_res) v += (float)resid[(size_t)pix * C_CH + co];
            v = v > 0.0f ? v : 0.0f;
            if (out_mode == 0) {
                outh[(size_t)pix * C_CH + co] = (bf16)v;
            } else {
                const int ob = pix >> 14;
                const int orem = pix & (HWp - 1);
                const size_t oidx = ((size_t)ob * C_CH + co) * HWp + orem;
                if (out_mode == 1) {
                    outf[oidx] = v;
                } else {
                    // keep "fine" (already in outf) where cmap > 0.5
                    if (!(cmap[pix] > 0.5f)) outf[oidx] = v;
                }
            }
        }
    }
}

// ---------------------------------------------------------------------------
// Depthwise 3x3 (groups=C), dilation 1, pad 1, ReLU. NHWC bf16 in/out.
// ---------------------------------------------------------------------------
__global__ __launch_bounds__(256)
void k_depthwise(const bf16* __restrict__ in, const float* __restrict__ wd,
                 const float* __restrict__ bd, bf16* __restrict__ out) {
    unsigned i = blockIdx.x * 256u + threadIdx.x;          // over NPIX*C
    const int c = i & (C_CH - 1);
    const unsigned pix = i >> 8;
    const int b = pix >> 14;
    const int rem = pix & (HWp - 1);
    const int h = rem >> 7;
    const int w = rem & (Wdim - 1);
    float acc = bd[c];
#pragma unroll
    for (int ky = 0; ky < 3; ++ky)
#pragma unroll
        for (int kx = 0; kx < 3; ++kx) {
            const int sh = h + ky - 1, sw = w + kx - 1;
            if (sh >= 0 && sh < Hdim && sw >= 0 && sw < Wdim) {
                float xv = (float)in[((size_t)(b * HWp + sh * Wdim + sw)) * C_CH + c];
                acc += xv * wd[c * 9 + ky * 3 + kx];
            }
        }
    acc = acc > 0.0f ? acc : 0.0f;
    out[i] = (bf16)acc;
}

// ---------------------------------------------------------------------------
extern "C" void kernel_launch(void* const* d_in, const int* in_sizes, int n_in,
                              void* d_out, int out_size, void* d_ws, size_t ws_size,
                              hipStream_t stream) {
    const float* x    = (const float*)d_in[0];
    const float* cmap = (const float*)d_in[1];
    const float* w1   = (const float*)d_in[2];
    const float* b1   = (const float*)d_in[3];
    const float* w2   = (const float*)d_in[4];
    const float* b2   = (const float*)d_in[5];
    const float* wd   = (const float*)d_in[6];
    const float* bd   = (const float*)d_in[7];
    const float* wp   = (const float*)d_in[8];
    const float* bp   = (const float*)d_in[9];
    float* out = (float*)d_out;

    char* ws = (char*)d_ws;
    bf16* xh  = (bf16*)(ws);                                // 64 MiB NHWC bf16
    bf16* t1  = (bf16*)(ws + (size_t)64 * 1024 * 1024);     // 64 MiB NHWC bf16
    bf16* w1p = (bf16*)(ws + (size_t)128 * 1024 * 1024);    // [9][256][256]
    bf16* w2p = (bf16*)(ws + (size_t)128 * 1024 * 1024 + (size_t)2 * 1024 * 1024);
    bf16* wpp = (bf16*)(ws + (size_t)128 * 1024 * 1024 + (size_t)4 * 1024 * 1024);

    const int nElem = NPIX * C_CH;                          // 33,554,432

    // Stage 0: layout conversion + weight packing
    k_nchw_to_nhwc<<<nElem / 256, 256, 0, stream>>>(x, xh);
    k_pack_w3<<<(9 * C_CH * C_CH) / 256, 256, 0, stream>>>(w1, w1p);
    k_pack_w3<<<(9 * C_CH * C_CH) / 256, 256, 0, stream>>>(w2, w2p);
    k_pack_w1<<<(C_CH * C_CH) / 256, 256, 0, stream>>>(wp, wpp);

    const int gemmBlocks = NPIX / 128;                      // 1024

    // Fine branch: residual dilated conv1 (d=2) -> t1 (bf16 NHWC)
    k_conv_gemm<<<gemmBlocks, 256, 0, stream>>>(
        xh, w1p, b1, /*resid=*/xh, /*cmap=*/nullptr,
        /*outf=*/nullptr, /*outh=*/t1, /*ntaps=*/9, /*dil=*/2, /*out_mode=*/0);

    // Fine branch: residual dilated conv2 (d=4) -> d_out (f32 NCHW, "fine")
    k_conv_gemm<<<gemmBlocks, 256, 0, stream>>>(
        t1, w2p, b2, /*resid=*/t1, /*cmap=*/nullptr,
        /*outf=*/out, /*outh=*/nullptr, /*ntaps=*/9, /*dil=*/4, /*out_mode=*/1);

    // Coarse branch: depthwise (xh -> t1, reusing buffer after conv2 consumed it)
    k_depthwise<<<nElem / 256, 256, 0, stream>>>(xh, wd, bd, t1);

    // Coarse pointwise GEMM + fused routing: overwrite d_out where cmap <= 0.5
    k_conv_gemm<<<gemmBlocks, 256, 0, stream>>>(
        t1, wpp, bp, /*resid=*/nullptr, cmap,
        /*outf=*/out, /*outh=*/nullptr, /*ntaps=*/1, /*dil=*/1, /*out_mode=*/2);
}
